// GTN_47794396070630
// MI455X (gfx1250) — matrix operand order repaired and checked by reading, not compiled, not run on hardware
//
#include <hip/hip_runtime.h>
#include <hip/hip_bf16.h>

// ---------------- problem constants ----------------
constexpr int kE = 5, kC = 2, kN = 2048, kWin = 512, kWout = 256;
constexpr long long kNN2 = (long long)kN * kN;

typedef __attribute__((ext_vector_type(16))) __bf16 bf16x16;
typedef __attribute__((ext_vector_type(8)))  __bf16 bf16x8;
typedef __attribute__((ext_vector_type(8)))  float  v8f;

// Async global->LDS (gfx1250). Guarded so host pass / older clang still compile.
#if __has_builtin(__builtin_amdgcn_global_load_async_to_lds_b128) && __has_builtin(__builtin_amdgcn_s_wait_asynccnt)
#define USE_ASYNC 1
#else
#define USE_ASYNC 0
#endif

#if USE_ASYNC
// builtin signature (from hipcc diagnostic): (v4i as1*, v4i as3*, imm offset, imm cpol)
typedef int v4i __attribute__((vector_size(16)));
using gv4i = __attribute__((address_space(1))) v4i;
using lv4i = __attribute__((address_space(3))) v4i;
__device__ __forceinline__ void async_cp16(const __bf16* g, __bf16* l) {
  __builtin_amdgcn_global_load_async_to_lds_b128(
      (gv4i*)(unsigned long long)(g),
      (lv4i*)(unsigned int)(unsigned long long)(l),
      0, 0);
}
#endif

// ---------------- small kernels ----------------

// softmax over E for the three (C,E) GTConv weight sets -> f3[set][c][e]
__global__ void k_softmax(const float* w0a, const float* w0b, const float* w1, float* f3) {
  int t = threadIdx.x;
  if (t >= 3 * kC) return;
  int set = t / kC, c = t % kC;
  const float* w = (set == 0) ? w0a : ((set == 1) ? w0b : w1);
  float m = -1e30f;
  for (int e = 0; e < kE; e++) m = fmaxf(m, w[c * kE + e]);
  float ex[kE], s = 0.f;
  for (int e = 0; e < kE; e++) { ex[e] = __expf(w[c * kE + e] - m); s += ex[e]; }
  for (int e = 0; e < kE; e++) f3[(set * kC + c) * kE + e] = ex[e] / s;
}

// one pass over A: Ha row-major bf16, Hb / Hb2 column-major bf16 (LDS transpose)
__global__ __launch_bounds__(256) void k_gtconv(const float* __restrict__ A,
                                                const float* __restrict__ f3,
                                                __bf16* __restrict__ Ha,
                                                __bf16* __restrict__ Hb,
                                                __bf16* __restrict__ Hb2) {
  __shared__ float fs[3 * kC * kE];
  __shared__ __bf16 shb[kC][32][33];
  __shared__ __bf16 shb2[kC][32][33];
  int tx = threadIdx.x, ty = threadIdx.y;
  int tid = ty * 32 + tx;
  if (tid < 3 * kC * kE) fs[tid] = f3[tid];
  __syncthreads();
  int n0 = blockIdx.y * 32, m0 = blockIdx.x * 32;
  for (int rr = 0; rr < 4; rr++) {
    int r = ty + rr * 8;
    int n = n0 + r, m = m0 + tx;
    float a[kE];
#pragma unroll
    for (int e = 0; e < kE; e++) a[e] = A[(long long)e * kNN2 + (long long)n * kN + m];
#pragma unroll
    for (int c = 0; c < kC; c++) {
      float ha = 0.f, hb = 0.f, h2 = 0.f;
#pragma unroll
      for (int e = 0; e < kE; e++) {
        ha += fs[(0 * kC + c) * kE + e] * a[e];
        hb += fs[(1 * kC + c) * kE + e] * a[e];
        h2 += fs[(2 * kC + c) * kE + e] * a[e];
      }
      Ha[(long long)c * kNN2 + (long long)n * kN + m] = (__bf16)ha;
      shb[c][r][tx]  = (__bf16)hb;
      shb2[c][r][tx] = (__bf16)h2;
    }
  }
  __syncthreads();
  for (int rr = 0; rr < 4; rr++) {
    int r = ty + rr * 8;
    int m = m0 + r, n = n0 + tx;
#pragma unroll
    for (int c = 0; c < kC; c++) {
      Hb[(long long)c * kNN2 + (long long)m * kN + n]  = shb[c][tx][r];
      Hb2[(long long)c * kNN2 + (long long)m * kN + n] = shb2[c][tx][r];
    }
  }
}

// column degrees (diag excluded) of f32 matrix -> dinv
__global__ void k_deg(const float* __restrict__ Cb, float* __restrict__ dinv) {
  int j = blockIdx.x * blockDim.x + threadIdx.x;
  int c = blockIdx.y;
  const float* p = Cb + (long long)c * kNN2;
  float s = 0.f;
  for (int i = 0; i < kN; i++) s += p[(long long)i * kN + j];
  s -= p[(long long)j * kN + j];
  dinv[c * kN + j] = (s != 0.f) ? 1.f / s : 0.f;
}

// H[i][j] = (i==j ? 0 : C[i][j]*dinv[j]) -> bf16 row-major
__global__ void k_scale_norm(const float* __restrict__ Cb, const float* __restrict__ dinv,
                             __bf16* __restrict__ H) {
  long long i = (long long)blockIdx.x * blockDim.x + threadIdx.x;
  if (i >= (long long)kC * kNN2) return;
  int c = (int)(i / kNN2);
  long long rem = i % kNN2;
  int r = (int)(rem / kN), j = (int)(rem % kN);
  float v = (r == j) ? 0.f : Cb[i] * dinv[c * kN + j];
  H[i] = (__bf16)v;
}

// H2T[j][i] = (i==j ? 0 : C[i][j]*dinv[j]) -> bf16, tiled transpose
__global__ __launch_bounds__(256) void k_tscale(const float* __restrict__ Cb,
                                                const float* __restrict__ dinv,
                                                __bf16* __restrict__ H2T) {
  __shared__ float sh[32][33];
  int tx = threadIdx.x, ty = threadIdx.y;
  int c = blockIdx.z;
  int i0 = blockIdx.x * 32, j0 = blockIdx.y * 32;
  const float* p = Cb + (long long)c * kNN2;
  for (int rr = 0; rr < 4; rr++) {
    int r = ty + rr * 8;
    sh[r][tx] = p[(long long)(i0 + r) * kN + (j0 + tx)];
  }
  __syncthreads();
  for (int rr = 0; rr < 4; rr++) {
    int r = ty + rr * 8;
    int j = j0 + r, i = i0 + tx;
    float v = (i == j) ? 0.f : sh[tx][r] * dinv[c * kN + j];
    H2T[(long long)c * kNN2 + (long long)j * kN + i] = (__bf16)v;
  }
}

// dis[c][n] = rsqrt(1 + row-sum of H2T[c][n][:])
__global__ void k_rowsum_dis(const __bf16* __restrict__ H2T, float* __restrict__ dis) {
  __shared__ float red[256];
  int n = blockIdx.x, c = blockIdx.y, tid = threadIdx.x;
  float s = 0.f;
  for (int r = tid; r < kN; r += 256)
    s += (float)H2T[(long long)c * kNN2 + (long long)n * kN + r];
  red[tid] = s;
  __syncthreads();
  for (int st = 128; st > 0; st >>= 1) {
    if (tid < st) red[tid] += red[tid + st];
    __syncthreads();
  }
  if (tid == 0) {
    float deg = 1.f + red[0];
    dis[c * kN + n] = (deg > 0.f) ? rsqrtf(deg) : 0.f;
  }
}

// XW = X (2048x512) @ gcn_w (512x256), f32, plain tiled
__global__ __launch_bounds__(256) void k_xw(const float* __restrict__ X,
                                            const float* __restrict__ W,
                                            float* __restrict__ XW) {
  __shared__ float sx[16][17], sw[16][17];
  int tx = threadIdx.x, ty = threadIdx.y;
  int f = blockIdx.x * 16 + tx, n = blockIdx.y * 16 + ty;
  float acc = 0.f;
  for (int k0 = 0; k0 < kWin; k0 += 16) {
    sx[ty][tx] = X[(long long)n * kWin + k0 + tx];
    sw[ty][tx] = W[(long long)(k0 + ty) * kWout + f];
    __syncthreads();
#pragma unroll
    for (int kk = 0; kk < 16; kk++) acc += sx[ty][kk] * sw[kk][tx];
    __syncthreads();
  }
  XW[(long long)n * kWout + f] = acc;
}

// XWs[c][f][r] = bf16(XW[r][f] * dis[c][r])  (col-major B operand)
__global__ void k_xws(const float* __restrict__ XW, const float* __restrict__ dis,
                      __bf16* __restrict__ XWs) {
  long long i = (long long)blockIdx.x * blockDim.x + threadIdx.x;
  if (i >= (long long)kC * kWout * kN) return;
  int c = (int)(i / ((long long)kWout * kN));
  long long rem = i % ((long long)kWout * kN);
  int f = (int)(rem / kN), r = (int)(rem % kN);
  XWs[i] = (__bf16)(XW[(long long)r * kWout + f] * dis[c * kN + r]);
}

__global__ void k_f2b(const float* __restrict__ s, __bf16* __restrict__ d, int n) {
  int i = blockIdx.x * blockDim.x + threadIdx.x;
  if (i < n) d[i] = (__bf16)s[i];
}

// ---------------- generic bf16 WMMA GEMM ----------------
// C[z] (MxN) = A[z] (MxK, row-major bf16) @ B[z] (KxN given as Bt: NxK row-major bf16)
// block tile 128x128, 8 waves of 32x64, K steps of 32, LDS double-buffered async.
constexpr int kTM = 128, kTN = 128, kLDS = 40; // 40 bf16 row stride (80B, conflict-padded)

union Frag { bf16x16 v; bf16x8 h[2]; };

__global__ __launch_bounds__(256) void k_gemm(
    const __bf16* __restrict__ Abase, long long sA, int lda,
    const __bf16* __restrict__ Bbase, long long sB, int ldb,
    int K, int mode,
    float* __restrict__ outF, long long sF, int ldF,
    __bf16* __restrict__ outH, int ldH, int colOffZ,
    const float* __restrict__ dis, int disStride,
    const float* __restrict__ XW, int ldXW,
    const float* __restrict__ bias) {
  __shared__ __align__(16) __bf16 shA[2][kTM * kLDS];
  __shared__ __align__(16) __bf16 shB[2][kTN * kLDS];
  const int z = blockIdx.z;
  const int m0 = blockIdx.y * kTM;
  const int n0 = blockIdx.x * kTN;
  const __bf16* A = Abase + (long long)z * sA;
  const __bf16* B = Bbase + (long long)z * sB;
  const int tid = threadIdx.x;
  const int lane = tid & 31, wid = tid >> 5;
  const int wm = wid >> 1, wn = wid & 1;   // 4x2 wave grid -> wave tile 32 rows x 64 cols
  const int l16 = lane & 15, lh = lane >> 4;

  v8f acc[2][4] = {};

  auto issue = [&](int buf, int k0) {
#pragma unroll
    for (int t = 0; t < 2; t++) {
      int c = tid + t * 256;            // 512 16B chunks per 128x32 tile
      int row = c >> 2, seg = c & 3;
      const __bf16* ga = A + (long long)(m0 + row) * lda + k0 + seg * 8;
      __bf16* la = &shA[buf][row * kLDS + seg * 8];
      const __bf16* gb = B + (long long)(n0 + row) * ldb + k0 + seg * 8;
      __bf16* lb = &shB[buf][row * kLDS + seg * 8];
#if USE_ASYNC
      async_cp16(ga, la);
      async_cp16(gb, lb);
#else
      *(float4*)la = *(const float4*)ga;
      *(float4*)lb = *(const float4*)gb;
#endif
    }
  };

  auto compute = [&](int buf) {
    Frag af[2], bfr[4];
#pragma unroll
    for (int mf = 0; mf < 2; mf++) {
      int row = wm * 32 + mf * 16 + l16;
      af[mf].h[0] = *(const bf16x8*)&shA[buf][row * kLDS + lh * 8];
      af[mf].h[1] = *(const bf16x8*)&shA[buf][row * kLDS + lh * 8 + 16];
    }
#pragma unroll
    for (int nf = 0; nf < 4; nf++) {
      int col = wn * 64 + nf * 16 + l16;
      bfr[nf].h[0] = *(const bf16x8*)&shB[buf][col * kLDS + lh * 8];
      bfr[nf].h[1] = *(const bf16x8*)&shB[buf][col * kLDS + lh * 8 + 16];
    }
#pragma unroll
    for (int mf = 0; mf < 2; mf++)
#pragma unroll
      for (int nf = 0; nf < 4; nf++)
        acc[mf][nf] = __builtin_amdgcn_wmma_f32_16x16x32_bf16(
            false, af[mf].v, false, bfr[nf].v, (short)0, acc[mf][nf], false, false);
  };

  const int S = K / 32;
#if USE_ASYNC
  issue(0, 0);
  for (int s = 0; s < S; s++) {
    if (s + 1 < S) {
      issue((s + 1) & 1, (s + 1) * 32);
      __builtin_amdgcn_s_wait_asynccnt(4);   // this wave issues 4 async ops per stage
    } else {
      __builtin_amdgcn_s_wait_asynccnt(0);
    }
    __syncthreads();
    compute(s & 1);
    __syncthreads();
  }
#else
  for (int s = 0; s < S; s++) {
    __syncthreads();
    issue(0, s * 32);
    __syncthreads();
    compute(0);
  }
#endif

  // epilogue: C/D layout — VGPR v, lane: row = v + (lane>=16)*8, col = lane%16
#pragma unroll
  for (int mf = 0; mf < 2; mf++)
#pragma unroll
    for (int nf = 0; nf < 4; nf++)
#pragma unroll
      for (int v = 0; v < 8; v++) {
        int row = m0 + wm * 32 + mf * 16 + lh * 8 + v;
        int col = n0 + wn * 64 + nf * 16 + l16;
        float val = acc[mf][nf][v];
        if (mode == 0) {
          outF[(long long)z * sF + (long long)row * ldF + col] = val;
        } else if (mode == 1) {  // GCN epilogue
          float d = dis[z * disStride + row];
          val = d * (val + d * XW[(long long)row * ldXW + col]) + bias[col];
          val = fmaxf(val, 0.f);
          outH[(long long)row * ldH + z * colOffZ + col] = (__bf16)val;
        } else if (mode == 2) {  // relu(acc + bias) -> bf16
          val = fmaxf(val + bias[col], 0.f);
          outH[(long long)row * ldH + col] = (__bf16)val;
        } else {                 // acc + bias -> f32
          outF[(long long)row * ldF + col] = val + bias[col];
        }
      }
}

// ---------------- host launcher ----------------
extern "C" void kernel_launch(void* const* d_in, const int* in_sizes, int n_in,
                              void* d_out, int out_size, void* d_ws, size_t ws_size,
                              hipStream_t stream) {
  const float* A      = (const float*)d_in[0];
  const float* X      = (const float*)d_in[1];
  const float* gt_w0a = (const float*)d_in[2];
  const float* gt_w0b = (const float*)d_in[3];
  const float* gt_w1  = (const float*)d_in[4];
  const float* gcn_w  = (const float*)d_in[5];
  const float* gcn_b  = (const float*)d_in[6];
  const float* lin1_w = (const float*)d_in[7];
  const float* lin1_b = (const float*)d_in[8];
  const float* lin2_w = (const float*)d_in[9];
  const float* lin2_b = (const float*)d_in[10];

  char* w = (char*)d_ws;
  size_t off = 0;
  auto alloc = [&](size_t bytes) -> char* {
    char* p = w + off;
    off += (bytes + 255) & ~(size_t)255;
    return p;
  };
  float*  f3    = (float*)alloc(3 * kC * kE * sizeof(float));
  __bf16* Ha    = (__bf16*)alloc((size_t)kC * kNN2 * 2);  // reused as H1
  __bf16* Hb    = (__bf16*)alloc((size_t)kC * kNN2 * 2);  // reused as H2T
  __bf16* Hb2   = (__bf16*)alloc((size_t)kC * kNN2 * 2);
  float*  Cbuf  = (float*)alloc((size_t)kC * kNN2 * 4);   // C1 then C2
  float*  dinv1 = (float*)alloc((size_t)kC * kN * 4);
  float*  dinv2 = (float*)alloc((size_t)kC * kN * 4);
  float*  dis   = (float*)alloc((size_t)kC * kN * 4);
  float*  XW    = (float*)alloc((size_t)kN * kWout * 4);
  __bf16* XWs   = (__bf16*)alloc((size_t)kC * kWout * kN * 2);
  __bf16* Xcat  = (__bf16*)alloc((size_t)kN * (kC * kWout) * 2);
  __bf16* hbuf  = (__bf16*)alloc((size_t)kN * kWout * 2);
  __bf16* w1b   = (__bf16*)alloc((size_t)kWout * kWin * 2);
  __bf16* w2b   = (__bf16*)alloc((size_t)kWout * kWout * 2);

  __bf16* H1  = Ha;  // alias: Ha dead after GEMM1
  __bf16* H2T = Hb;  // alias: Hb dead after GEMM1

  // 1. softmax of GTConv weights
  k_softmax<<<1, 32, 0, stream>>>(gt_w0a, gt_w0b, gt_w1, f3);
  // 2. weighted adjacency sums -> Ha (rm), Hb/Hb2 (cm)
  k_gtconv<<<dim3(kN / 32, kN / 32), dim3(32, 8), 0, stream>>>(A, f3, Ha, Hb, Hb2);
  // 3. C1 = Ha @ Hb (per channel)
  k_gemm<<<dim3(kN / kTN, kN / kTM, kC), 256, 0, stream>>>(
      Ha, kNN2, kN, Hb, kNN2, kN, kN, 0,
      Cbuf, kNN2, kN, nullptr, 0, 0, nullptr, 0, nullptr, 0, nullptr);
  // 4-5. normalize -> H1 (bf16)
  k_deg<<<dim3(kN / 256, kC), 256, 0, stream>>>(Cbuf, dinv1);
  k_scale_norm<<<(unsigned)((kC * kNN2 + 255) / 256), 256, 0, stream>>>(Cbuf, dinv1, H1);
  // 6. C2 = H1 @ Hb2
  k_gemm<<<dim3(kN / kTN, kN / kTM, kC), 256, 0, stream>>>(
      H1, kNN2, kN, Hb2, kNN2, kN, kN, 0,
      Cbuf, kNN2, kN, nullptr, 0, 0, nullptr, 0, nullptr, 0, nullptr);
  // 7-8. normalize + transpose -> H2T (bf16)
  k_deg<<<dim3(kN / 256, kC), 256, 0, stream>>>(Cbuf, dinv2);
  k_tscale<<<dim3(kN / 32, kN / 32, kC), dim3(32, 8), 0, stream>>>(Cbuf, dinv2, H2T);
  // 9. GCN degrees -> dis
  k_rowsum_dis<<<dim3(kN, kC), 256, 0, stream>>>(H2T, dis);
  // 10-12. XW, scaled XW^T per channel, bf16 weights
  k_xw<<<dim3(kWout / 16, kN / 16), dim3(16, 16), 0, stream>>>(X, gcn_w, XW);
  k_xws<<<(unsigned)(((long long)kC * kWout * kN + 255) / 256), 256, 0, stream>>>(XW, dis, XWs);
  k_f2b<<<(kWout * kWin + 255) / 256, 256, 0, stream>>>(lin1_w, w1b, kWout * kWin);
  k_f2b<<<(kWout * kWout + 255) / 256, 256, 0, stream>>>(lin2_w, w2b, kWout * kWout);
  // 13. Xcat = relu(dis*(H2T@XWs + dis*XW) + gcn_b), channels concat in columns
  k_gemm<<<dim3(kWout / kTN, kN / kTM, kC), 256, 0, stream>>>(
      H2T, kNN2, kN, XWs, (long long)kWout * kN, kN, kN, 1,
      nullptr, 0, 0, Xcat, kC * kWout, kWout, dis, kN, XW, kWout, gcn_b);
  // 14. h = relu(Xcat @ lin1_w^T + lin1_b)
  k_gemm<<<dim3(kWout / kTN, kN / kTM, 1), 256, 0, stream>>>(
      Xcat, 0, kC * kWout, w1b, 0, kWin, kWin, 2,
      nullptr, 0, 0, hbuf, kWout, 0, nullptr, 0, nullptr, 0, lin1_b);
  // 15. out = h @ lin2_w^T + lin2_b  (f32 -> d_out)
  k_gemm<<<dim3(kWout / kTN, kN / kTM, 1), 256, 0, stream>>>(
      hbuf, 0, kWout, w2b, 0, kWout, kWout, 3,
      (float*)d_out, 0, kWout, nullptr, 0, 0, nullptr, 0, nullptr, 0, lin2_b);

  (void)in_sizes; (void)n_in; (void)out_size; (void)ws_size;
}
